// _PointMAESelfAttention_84447646974272
// MI455X (gfx1250) — matrix operand order repaired and checked by reading, not compiled, not run on hardware
//
#include <hip/hip_runtime.h>
#include <hip/hip_bf16.h>

// ---------------------------------------------------------------------------
// PointMAE self-attention for MI455X (gfx1250), bf16 WMMA + flash attention.
//   B=8, T=1024, DIM=512, NUM_HEADS=8, HEAD_DIM=64, SCALE=1/8
// Pipeline:
//   cvt(x,qkv_w,proj_w) -> qkv_gemm -> flash_attn -> proj_gemm
// flash_attn stages K/V tiles in LDS via the CDNA5 async load-to-LDS path
// (ASYNCcnt), removing the 8x redundant K/V global traffic per workgroup.
// ---------------------------------------------------------------------------

typedef __bf16 bf16_t;
typedef __attribute__((ext_vector_type(16))) __bf16 v16bf;
typedef __attribute__((ext_vector_type(8)))  __bf16 v8bf;
typedef __attribute__((ext_vector_type(4)))  __bf16 v4bf;
typedef __attribute__((ext_vector_type(8)))  float  v8f;
typedef __attribute__((ext_vector_type(4)))  float  v4f;
typedef __attribute__((ext_vector_type(4)))  int    v4i;

#define DIMC   512
#define NHEAD  8
#define HDIM   64
#define BATCH  8
#define SEQT   1024
#define MTOT   (BATCH * SEQT)       /* 8192 */
#define ATTN_SCALE 0.125f
#define MASK_MIN  (-3.402823466e38f)

#if defined(__has_builtin)
#if __has_builtin(__builtin_amdgcn_global_load_async_to_lds_b128)
#define HAVE_ASYNC_LDS 1
#endif
#endif

__device__ __forceinline__ v8f wmma_bf16(v16bf a, v16bf b, v8f c) {
  // D = A(16x32 bf16) * B(32x16 bf16) + C(16x16 f32)
  return __builtin_amdgcn_wmma_f32_16x16x32_bf16(
      /*neg_a=*/false, a, /*neg_b=*/false, b,
      /*c_mod=*/(short)0, c, /*reuse_a=*/false, /*reuse_b=*/false);
}

// Fragment = concat of two contiguous 8-element bf16 chunks (ISA layout:
// A- and B-frag chunks are both 16 elements apart).
__device__ __forceinline__ v16bf load_frag(const bf16_t* p0, const bf16_t* p1) {
  v8bf c0 = *(const v8bf*)p0;
  v8bf c1 = *(const v8bf*)p1;
  v16bf r;
#pragma unroll
  for (int i = 0; i < 8; ++i) { r[i] = c0[i]; r[i + 8] = c1[i]; }
  return r;
}

// 16-byte global -> LDS stage; async (ASYNCcnt) when the toolchain has the
// gfx1250 builtin, otherwise a register bounce.
__device__ __forceinline__ void stage16(const bf16_t* g, bf16_t* l) {
#ifdef HAVE_ASYNC_LDS
  __builtin_amdgcn_global_load_async_to_lds_b128(
      (v4i*)g, (v4i*)l, /*offset=*/0, /*cpol=*/0);
#else
  *(v8bf*)l = *(const v8bf*)g;
#endif
}

__device__ __forceinline__ void stage_drain() {
#ifdef HAVE_ASYNC_LDS
  asm volatile("s_wait_asynccnt 0" ::: "memory");
#endif
}

// ---------------------------------------------------------------------------
// Kernel 0: fp32 -> bf16 conversion, 4 elements / thread
// ---------------------------------------------------------------------------
__global__ void cvt_f32_bf16_x4(const float* __restrict__ in,
                                bf16_t* __restrict__ out, int n4) {
  int i = blockIdx.x * blockDim.x + threadIdx.x;
  if (i >= n4) return;
  v4f f = *(const v4f*)(in + (size_t)i * 4);
  v4bf b;
#pragma unroll
  for (int k = 0; k < 4; ++k) b[k] = (bf16_t)f[k];
  *(v4bf*)(out + (size_t)i * 4) = b;
}

// ---------------------------------------------------------------------------
// Kernel 1: qkv = x @ qkv_w^T + qkv_b  -> scatter to Q[b,h,t,d], K[b,h,t,d],
//           V^T[b,h,d,t] (bf16). Block tile 128x64, wave tile 32x32.
// ---------------------------------------------------------------------------
__global__ __launch_bounds__(256) void qkv_gemm(
    const bf16_t* __restrict__ xb, const bf16_t* __restrict__ wb,
    const float* __restrict__ bias, bf16_t* __restrict__ qb,
    bf16_t* __restrict__ kb, bf16_t* __restrict__ vtb) {
  const int lane = threadIdx.x & 31;
  const int w    = threadIdx.x >> 5;
  const int l16  = lane & 15;
  const int hl   = lane >> 4;
  const int mbase = blockIdx.y * 128 + (w & 3) * 32;
  const int nbase = blockIdx.x * 64  + (w >> 2) * 32;

  v8f acc[2][2] = {};

  const bf16_t* arow0 = xb + (size_t)(mbase + l16) * DIMC;
  const bf16_t* arow1 = arow0 + 16 * DIMC;
  const bf16_t* brow0 = wb + (size_t)(nbase + l16) * DIMC;
  const bf16_t* brow1 = brow0 + 16 * DIMC;

#pragma unroll 4
  for (int kk = 0; kk < DIMC; kk += 32) {
    v16bf a0 = load_frag(arow0 + kk + hl * 8,  arow0 + kk + hl * 8 + 16);
    v16bf a1 = load_frag(arow1 + kk + hl * 8,  arow1 + kk + hl * 8 + 16);
    v16bf b0 = load_frag(brow0 + kk + hl * 8, brow0 + kk + hl * 8 + 16);
    v16bf b1 = load_frag(brow1 + kk + hl * 8, brow1 + kk + hl * 8 + 16);
    acc[0][0] = wmma_bf16(a0, b0, acc[0][0]);
    acc[0][1] = wmma_bf16(a0, b1, acc[0][1]);
    acc[1][0] = wmma_bf16(a1, b0, acc[1][0]);
    acc[1][1] = wmma_bf16(a1, b1, acc[1][1]);
  }

#pragma unroll
  for (int fm = 0; fm < 2; ++fm)
#pragma unroll
    for (int fn = 0; fn < 2; ++fn)
#pragma unroll
      for (int r = 0; r < 8; ++r) {
        int m = mbase + fm * 16 + r + hl * 8;       // C layout: row r + 8*half
        int n = nbase + fn * 16 + l16;              //           col lane&15
        bf16_t bv = (bf16_t)(acc[fm][fn][r] + bias[n]);
        int which = n >> 9;                         // 0=q 1=k 2=v
        int c = n & 511, h = c >> 6, d = c & 63;
        int b = m >> 10, t = m & 1023;
        size_t bh = (size_t)(b * NHEAD + h);
        if (which == 0)      qb [(bh * SEQT + t) * HDIM + d] = bv;
        else if (which == 1) kb [(bh * SEQT + t) * HDIM + d] = bv;
        else                 vtb[(bh * HDIM + d) * SEQT + t] = bv;
      }
}

// ---------------------------------------------------------------------------
// Kernel 2: flash attention. grid=(T/128, B*H), 8 waves; each wave owns 16
// q-rows. K/V tiles for each 32-column chunk are staged once per workgroup
// into LDS (async load-to-LDS), then all waves build B-fragments from LDS.
// ---------------------------------------------------------------------------
__global__ __launch_bounds__(256) void flash_attn(
    const bf16_t* __restrict__ qb, const bf16_t* __restrict__ kb,
    const bf16_t* __restrict__ vtb, const unsigned char* __restrict__ mask,
    bf16_t* __restrict__ ao) {
  const int tid  = threadIdx.x;
  const int lane = tid & 31;
  const int w    = tid >> 5;
  const int l16  = lane & 15;
  const int hl   = lane >> 4;
  const int bh   = blockIdx.y;            // b*8 + h
  const int bidx = bh >> 3, hidx = bh & 7;
  const int q0   = blockIdx.x * 128 + w * 16;

  const bf16_t* Q  = qb  + (size_t)bh * SEQT * HDIM;
  const bf16_t* K  = kb  + (size_t)bh * SEQT * HDIM;
  const bf16_t* Vt = vtb + (size_t)bh * HDIM * SEQT;
  const unsigned char* Mb = mask + (size_t)bidx * SEQT * SEQT;

  // Q A-fragments over the 64-wide head dim (two K=32 fragments)
  const bf16_t* qrow = Q + (size_t)(q0 + l16) * HDIM;
  v16bf qf0 = load_frag(qrow + hl * 8,       qrow + hl * 8 + 16);
  v16bf qf1 = load_frag(qrow + 32 + hl * 8,  qrow + 32 + hl * 8 + 16);

  float mrow[8], lrow[8];
  v8f acc[4] = {};
#pragma unroll
  for (int r = 0; r < 8; ++r) { mrow[r] = -__builtin_inff(); lrow[r] = 0.f; }

  // Shared tiles: K rows [j..j+32) x 64, V^T rows [0..64) x cols [j..j+32),
  // plus per-wave P staging (C-layout -> A-layout transpose).
  __shared__ __align__(16) bf16_t kt[32][64];        // 4 KB
  __shared__ __align__(16) bf16_t vt[64][32];        // 4 KB
  __shared__ __align__(16) bf16_t pbuf[8][16][32];   // 8 KB
  bf16_t (*pw)[32] = pbuf[w];

  // Cooperative staging map: one b128 per thread per tile.
  const int krow_s = tid >> 3, kseg = (tid & 7) * 8;   // 32 rows x 8 segs
  const int vrow_s = tid >> 2, vseg = (tid & 3) * 8;   // 64 rows x 4 segs

  for (int j = 0; j < SEQT; j += 32) {
    // ---- stage K[j..j+32)[0..64) and Vt[0..64)[j..j+32) into LDS ----
    stage16(K  + (size_t)(j + krow_s) * HDIM + kseg, &kt[krow_s][kseg]);
    stage16(Vt + (size_t)vrow_s * SEQT + j + vseg,   &vt[vrow_s][vseg]);
    stage_drain();                 // wave's ASYNCcnt -> 0
    __syncthreads();               // tiles visible to all waves

    // ---- S = Q @ K^T for 32 key columns (two 16-col fragments) ----
    v16bf kb0a = load_frag(&kt[l16][hl * 8],           &kt[l16][hl * 8 + 16]);
    v16bf kb0b = load_frag(&kt[l16][32 + hl * 8],      &kt[l16][32 + hl * 8 + 16]);
    v16bf kb1a = load_frag(&kt[16 + l16][hl * 8],      &kt[16 + l16][hl * 8 + 16]);
    v16bf kb1b = load_frag(&kt[16 + l16][32 + hl * 8], &kt[16 + l16][32 + hl * 8 + 16]);
    v8f s0 = {}, s1 = {};
    s0 = wmma_bf16(qf0, kb0a, s0); s0 = wmma_bf16(qf1, kb0b, s0);
    s1 = wmma_bf16(qf0, kb1a, s1); s1 = wmma_bf16(qf1, kb1b, s1);

    // ---- scale + mask (mask true -> finfo.min, matching reference) ----
    float v0[8], v1[8];
#pragma unroll
    for (int r = 0; r < 8; ++r) {
      size_t mi = (size_t)(q0 + r + 8 * hl) * SEQT + j + l16;
      v0[r] = Mb[mi]      ? MASK_MIN : s0[r] * ATTN_SCALE;
      v1[r] = Mb[mi + 16] ? MASK_MIN : s1[r] * ATTN_SCALE;
    }

    // ---- online softmax: row reductions across the 16-lane half ----
    float cm[8];
#pragma unroll
    for (int r = 0; r < 8; ++r) cm[r] = fmaxf(v0[r], v1[r]);
#pragma unroll
    for (int off = 1; off < 16; off <<= 1)
#pragma unroll
      for (int r = 0; r < 8; ++r)
        cm[r] = fmaxf(cm[r], __shfl_xor(cm[r], off, 32));

    float cf[8];
#pragma unroll
    for (int r = 0; r < 8; ++r) {
      float mn = fmaxf(mrow[r], cm[r]);
      cf[r] = __expf(mrow[r] - mn);      // exp(-inf)=0 on first chunk
      mrow[r] = mn;
      v0[r] = __expf(v0[r] - mn);
      v1[r] = __expf(v1[r] - mn);
    }
    float rs[8];
#pragma unroll
    for (int r = 0; r < 8; ++r) rs[r] = v0[r] + v1[r];
#pragma unroll
    for (int off = 1; off < 16; off <<= 1)
#pragma unroll
      for (int r = 0; r < 8; ++r) rs[r] += __shfl_xor(rs[r], off, 32);
#pragma unroll
    for (int r = 0; r < 8; ++r) lrow[r] = lrow[r] * cf[r] + rs[r];
#pragma unroll
    for (int f = 0; f < 4; ++f)
#pragma unroll
      for (int r = 0; r < 8; ++r) acc[f][r] *= cf[r];

    // ---- P: C-layout -> LDS -> A-layout fragment (wave-private region) ----
#pragma unroll
    for (int r = 0; r < 8; ++r) {
      pw[r + 8 * hl][l16]      = (bf16_t)v0[r];
      pw[r + 8 * hl][16 + l16] = (bf16_t)v1[r];
    }
    asm volatile("s_wait_dscnt 0" ::: "memory");   // wave-local LDS fence
    v16bf pa = load_frag(&pw[l16][hl * 8], &pw[l16][16 + hl * 8]);

    // ---- O += P @ V  (V^T tile in LDS: contiguous B-fragments) ----
#pragma unroll
    for (int f = 0; f < 4; ++f) {
      v16bf vb = load_frag(&vt[f * 16 + l16][hl * 8], &vt[f * 16 + l16][hl * 8 + 16]);
      acc[f] = wmma_bf16(pa, vb, acc[f]);
    }
    __syncthreads();               // before next chunk overwrites kt/vt
  }

  // ---- normalize and emit bf16 attn-out in [b, t, h*64+d] layout ----
#pragma unroll
  for (int f = 0; f < 4; ++f)
#pragma unroll
    for (int r = 0; r < 8; ++r) {
      int t = q0 + r + 8 * hl;
      int d = f * 16 + l16;
      float o = acc[f][r] / lrow[r];
      ao[((size_t)bidx * SEQT + t) * DIMC + hidx * HDIM + d] = (bf16_t)o;
    }
}

// ---------------------------------------------------------------------------
// Kernel 3: out = attn_out @ proj_w^T + proj_b  (fp32 epilogue)
// ---------------------------------------------------------------------------
__global__ __launch_bounds__(256) void proj_gemm(
    const bf16_t* __restrict__ ab, const bf16_t* __restrict__ wb,
    const float* __restrict__ bias, float* __restrict__ out) {
  const int lane = threadIdx.x & 31;
  const int w    = threadIdx.x >> 5;
  const int l16  = lane & 15;
  const int hl   = lane >> 4;
  const int mbase = blockIdx.y * 128 + (w & 3) * 32;
  const int nbase = blockIdx.x * 64  + (w >> 2) * 32;

  v8f acc[2][2] = {};

  const bf16_t* arow0 = ab + (size_t)(mbase + l16) * DIMC;
  const bf16_t* arow1 = arow0 + 16 * DIMC;
  const bf16_t* brow0 = wb + (size_t)(nbase + l16) * DIMC;
  const bf16_t* brow1 = brow0 + 16 * DIMC;

#pragma unroll 4
  for (int kk = 0; kk < DIMC; kk += 32) {
    v16bf a0 = load_frag(arow0 + kk + hl * 8,  arow0 + kk + hl * 8 + 16);
    v16bf a1 = load_frag(arow1 + kk + hl * 8,  arow1 + kk + hl * 8 + 16);
    v16bf b0 = load_frag(brow0 + kk + hl * 8, brow0 + kk + hl * 8 + 16);
    v16bf b1 = load_frag(brow1 + kk + hl * 8, brow1 + kk + hl * 8 + 16);
    acc[0][0] = wmma_bf16(a0, b0, acc[0][0]);
    acc[0][1] = wmma_bf16(a0, b1, acc[0][1]);
    acc[1][0] = wmma_bf16(a1, b0, acc[1][0]);
    acc[1][1] = wmma_bf16(a1, b1, acc[1][1]);
  }

#pragma unroll
  for (int fm = 0; fm < 2; ++fm)
#pragma unroll
    for (int fn = 0; fn < 2; ++fn)
#pragma unroll
      for (int r = 0; r < 8; ++r) {
        int m = mbase + fm * 16 + r + hl * 8;
        int n = nbase + fn * 16 + l16;
        out[(size_t)m * DIMC + n] = acc[fm][fn][r] + bias[n];
      }
}

// ---------------------------------------------------------------------------
extern "C" void kernel_launch(void* const* d_in, const int* in_sizes, int n_in,
                              void* d_out, int out_size, void* d_ws, size_t ws_size,
                              hipStream_t stream) {
  const float*         x      = (const float*)d_in[0];
  const unsigned char* mask   = (const unsigned char*)d_in[1]; // numpy bool
  const float*         qkv_w  = (const float*)d_in[2];
  const float*         qkv_b  = (const float*)d_in[3];
  const float*         proj_w = (const float*)d_in[4];
  const float*         proj_b = (const float*)d_in[5];
  float*               out    = (float*)d_out;

  bf16_t* xb   = (bf16_t*)d_ws;
  bf16_t* wqkv = xb   + (size_t)MTOT * DIMC;          // 8192*512
  bf16_t* pwb  = wqkv + (size_t)3 * DIMC * DIMC;      // 1536*512
  bf16_t* qb   = pwb  + (size_t)DIMC * DIMC;          // 512*512
  bf16_t* kbuf = qb   + (size_t)BATCH * NHEAD * SEQT * HDIM;
  bf16_t* vtb  = kbuf + (size_t)BATCH * NHEAD * SEQT * HDIM;
  bf16_t* ao   = vtb  + (size_t)BATCH * NHEAD * SEQT * HDIM;

  const int n4x = MTOT * DIMC / 4;
  const int n4w = 3 * DIMC * DIMC / 4;
  const int n4p = DIMC * DIMC / 4;
  cvt_f32_bf16_x4<<<(n4x + 255) / 256, 256, 0, stream>>>(x, xb, n4x);
  cvt_f32_bf16_x4<<<(n4w + 255) / 256, 256, 0, stream>>>(qkv_w, wqkv, n4w);
  cvt_f32_bf16_x4<<<(n4p + 255) / 256, 256, 0, stream>>>(proj_w, pwb, n4p);

  qkv_gemm<<<dim3(3 * DIMC / 64, MTOT / 128), 256, 0, stream>>>(
      xb, wqkv, qkv_b, qb, kbuf, vtb);

  flash_attn<<<dim3(SEQT / 128, BATCH * NHEAD), 256, 0, stream>>>(
      qb, kbuf, vtb, mask, ao);

  proj_gemm<<<dim3(DIMC / 64, MTOT / 128), 256, 0, stream>>>(
      ao, pwb, proj_b, out);
}
